// MultiHeadAttention_43834436223598
// MI455X (gfx1250) — matrix operand order repaired and checked by reading, not compiled, not run on hardware
//
#include <hip/hip_runtime.h>
#include <hip/hip_bf16.h>

// Problem constants (match reference)
#define BB 2
#define SS 2048
#define DD 1024
#define HH 16
#define DK 64           // DD / HH
#define BS (BB * SS)    // 4096 rows for projections

typedef __attribute__((ext_vector_type(16))) __bf16 v16bf;
typedef __attribute__((ext_vector_type(8)))  __bf16 v8bf;
typedef __attribute__((ext_vector_type(8)))  float  v8f;

static __device__ __forceinline__ __bf16 f2bf(float f) {
    unsigned u = __builtin_bit_cast(unsigned, f);
    unsigned r = u + 0x7FFFu + ((u >> 16) & 1u);   // round-to-nearest-even
    unsigned short h = (unsigned short)(r >> 16);
    return __builtin_bit_cast(__bf16, h);
}

static __device__ __forceinline__ v16bf cat8(v8bf lo, v8bf hi) {
    union { v16bf v; v8bf h[2]; } u;
    u.h[0] = lo; u.h[1] = hi;
    return u.v;
}

// ---------------------------------------------------------------------------
// fp32 -> bf16 conversion
// ---------------------------------------------------------------------------
__global__ void cvt_f32_to_bf16(const float* __restrict__ src,
                                __bf16* __restrict__ dst, int n) {
    int i = blockIdx.x * blockDim.x + threadIdx.x;
    if (i < n) dst[i] = f2bf(src[i]);
}

// ---------------------------------------------------------------------------
// Core 32x64 GEMM tile per wave:  C[32x64] += X[32 x DD] * W^T[DD x 64]
// X row-major [BS x DD], W row-major [DD x DD] (B column n = W row n).
// 8 WMMAs per K-step of 32; A fragments reused 4x, B fragments reused 2x.
// c[i][t]: i = M subtile (rows m0+16i), t = N subtile (cols n0+16t).
// ---------------------------------------------------------------------------
static __device__ __forceinline__ void gemm_tile_32x64(
        const __bf16* __restrict__ X, const __bf16* __restrict__ W,
        int m0, int n0, int lane, v8f c[2][4]) {
    const int half = lane >> 4;     // 0 or 1
    const int l16  = lane & 15;

    const __bf16* a0 = X + (size_t)(m0 + l16) * DD;
    const __bf16* a1 = X + (size_t)(m0 + 16 + l16) * DD;
    const __bf16* b0 = W + (size_t)(n0 + l16) * DD;
    const __bf16* b1 = W + (size_t)(n0 + 16 + l16) * DD;
    const __bf16* b2 = W + (size_t)(n0 + 32 + l16) * DD;
    const __bf16* b3 = W + (size_t)(n0 + 48 + l16) * DD;

    // warm the L2/WGP$ streams (global_prefetch_b8)
    __builtin_prefetch(a0, 0, 1);
    __builtin_prefetch(a1, 0, 1);
    __builtin_prefetch(b0, 0, 1);
    __builtin_prefetch(b2, 0, 1);

    for (int k0 = 0; k0 < DD; k0 += 32) {
        v16bf A[2];
        A[0] = cat8(*(const v8bf*)(a0 + k0 + half * 16),
                    *(const v8bf*)(a0 + k0 + half * 16 + 8));
        A[1] = cat8(*(const v8bf*)(a1 + k0 + half * 16),
                    *(const v8bf*)(a1 + k0 + half * 16 + 8));
        v16bf Bt[4];
        Bt[0] = *(const v16bf*)(b0 + k0 + half * 16);
        Bt[1] = *(const v16bf*)(b1 + k0 + half * 16);
        Bt[2] = *(const v16bf*)(b2 + k0 + half * 16);
        Bt[3] = *(const v16bf*)(b3 + k0 + half * 16);
#pragma unroll
        for (int t = 0; t < 4; ++t)
#pragma unroll
            for (int i = 0; i < 2; ++i)
                c[i][t] = __builtin_amdgcn_wmma_f32_16x16x32_bf16(
                    false, A[i], false, Bt[t], (short)0, c[i][t], false, false);
    }
}

// ---------------------------------------------------------------------------
// Projection GEMM:  out = X * W^T + bias  (bf16 out)
// vlayout == 0 : write [B, H, S, DK]   (for Q, K)
// vlayout == 1 : write [B, H, DK, S]   (for V, transposed for PV B-operand)
// ---------------------------------------------------------------------------
__global__ void proj_gemm_kernel(const __bf16* __restrict__ X,
                                 const __bf16* __restrict__ W,
                                 const float*  __restrict__ bias,
                                 __bf16* __restrict__ out, int vlayout) {
    const int lane = threadIdx.x & 31;
    const int wave = blockIdx.x * (blockDim.x >> 5) + (threadIdx.x >> 5);
    const int NT = DD / 64;                 // 16 column tiles of 64
    const int mt = wave / NT, nt = wave % NT;
    const int m0 = mt * 32, n0 = nt * 64;
    const int half = lane >> 4;
    const int l16  = lane & 15;

    v8f c[2][4] = {};
    gemm_tile_32x64(X, W, m0, n0, lane, c);

#pragma unroll
    for (int t = 0; t < 4; ++t) {
        const int n = n0 + t * 16 + l16;
        const float bcol = bias[n];
        const int hidx = n / DK, dk = n % DK;
#pragma unroll
        for (int i = 0; i < 2; ++i) {
#pragma unroll
            for (int j = 0; j < 8; ++j) {
                int r = m0 + i * 16 + (half ? 8 + j : j);
                int bidx = r / SS, srow = r % SS;
                float v = c[i][t][j] + bcol;
                size_t o;
                if (vlayout == 0)
                    o = (((size_t)(bidx * HH + hidx)) * SS + srow) * DK + dk;
                else
                    o = (((size_t)(bidx * HH + hidx)) * DK + dk) * SS + srow;
                out[o] = f2bf(v);
            }
        }
    }
}

// ---------------------------------------------------------------------------
// Output GEMM:  out(f32) = attn * Wo^T + bo   (row-major f32 output)
// ---------------------------------------------------------------------------
__global__ void out_gemm_kernel(const __bf16* __restrict__ X,
                                const __bf16* __restrict__ W,
                                const float*  __restrict__ bias,
                                float* __restrict__ out) {
    const int lane = threadIdx.x & 31;
    const int wave = blockIdx.x * (blockDim.x >> 5) + (threadIdx.x >> 5);
    const int NT = DD / 64;
    const int mt = wave / NT, nt = wave % NT;
    const int m0 = mt * 32, n0 = nt * 64;
    const int half = lane >> 4;
    const int l16  = lane & 15;

    v8f c[2][4] = {};
    gemm_tile_32x64(X, W, m0, n0, lane, c);

#pragma unroll
    for (int t = 0; t < 4; ++t) {
        const int n = n0 + t * 16 + l16;
        const float bcol = bias[n];
#pragma unroll
        for (int i = 0; i < 2; ++i) {
#pragma unroll
            for (int j = 0; j < 8; ++j) {
                int r = m0 + i * 16 + (half ? 8 + j : j);
                out[(size_t)r * DD + n] = c[i][t][j] + bcol;
            }
        }
    }
}

// ---------------------------------------------------------------------------
// Flash attention (causal). One wave per (b, h, 16-row q tile).
// Keys processed in chunks of 32; scores = 4 WMMAs, PV = 4 WMMAs per chunk.
// Online softmax with 16-lane shuffle reductions (row data lives in half-waves).
// P is re-laid-out C-layout -> A-layout via a 1KB/wave LDS staging buffer.
// ---------------------------------------------------------------------------
__global__ void attn_kernel(const __bf16* __restrict__ Qh,
                            const __bf16* __restrict__ Kh,
                            const __bf16* __restrict__ Vt,
                            __bf16* __restrict__ attn) {
    __shared__ __attribute__((aligned(32))) __bf16 pbuf[4][16 * 32];

    const int lane  = threadIdx.x & 31;
    const int wslot = threadIdx.x >> 5;
    const int wave  = blockIdx.x * 4 + wslot;
    const int QT = SS / 16;                       // 128
    const int qt = wave % QT;
    const int h  = (wave / QT) % HH;
    const int b  = wave / (QT * HH);
    const int q0 = qt * 16;
    const int half = lane >> 4, l16 = lane & 15;

    const size_t bh = (size_t)(b * HH + h);
    const __bf16* Qb = Qh + bh * SS * DK;
    const __bf16* Kb = Kh + bh * SS * DK;
    const __bf16* Vb = Vt + bh * DK * SS;         // [DK][S]

    // Q tile in A layout, K-dim 64 split into two 32-wide WMMA steps
    v16bf qa[2];
    {
        const __bf16* qrow = Qb + (size_t)(q0 + l16) * DK;
#pragma unroll
        for (int kk = 0; kk < 2; ++kk) {
            v8bf lo = *(const v8bf*)(qrow + kk * 32 + half * 16);
            v8bf hi = *(const v8bf*)(qrow + kk * 32 + half * 16 + 8);
            qa[kk] = cat8(lo, hi);
        }
    }

    v8f acc[4] = {};
    float mrow[8], lrow[8];
#pragma unroll
    for (int j = 0; j < 8; ++j) { mrow[j] = -1e30f; lrow[j] = 0.f; }

    const int nch = (q0 + 16 + 31) >> 5;          // causal chunk count
    for (int ch = 0; ch < nch; ++ch) {
        const int kc0 = ch * 32;

        // ---- scores: two 16-col tiles, each K=64 -> 2 WMMAs
        v8f sc[2];
#pragma unroll
        for (int nt2 = 0; nt2 < 2; ++nt2) {
            const __bf16* krow = Kb + (size_t)(kc0 + nt2 * 16 + l16) * DK;
            v8f s = {};
#pragma unroll
            for (int kk = 0; kk < 2; ++kk) {
                v16bf bm = *(const v16bf*)(krow + kk * 32 + half * 16);
                s = __builtin_amdgcn_wmma_f32_16x16x32_bf16(false, qa[kk], false,
                                                            bm, (short)0, s,
                                                            false, false);
            }
            sc[nt2] = s;
        }

        // ---- scale + causal mask (per-lane select, no EXEC divergence)
        const float scale = 0.125f;               // 1/sqrt(64)
#pragma unroll
        for (int nt2 = 0; nt2 < 2; ++nt2) {
            int col = kc0 + nt2 * 16 + l16;
#pragma unroll
            for (int j = 0; j < 8; ++j) {
                int row = q0 + (half ? 8 + j : j);
                float v = sc[nt2][j] * scale;
                sc[nt2][j] = (col > row) ? -1e30f : v;
            }
        }

        // ---- online softmax (reduce over 16 lanes of each half-wave)
#pragma unroll
        for (int j = 0; j < 8; ++j) {
            float tmax = fmaxf(sc[0][j], sc[1][j]);
            for (int d = 1; d < 16; d <<= 1)
                tmax = fmaxf(tmax, __shfl_xor(tmax, d, 32));
            float mnew = fmaxf(mrow[j], tmax);
            float p0 = __expf(sc[0][j] - mnew);
            float p1 = __expf(sc[1][j] - mnew);
            float rs = p0 + p1;
            for (int d = 1; d < 16; d <<= 1)
                rs += __shfl_xor(rs, d, 32);
            float corr = __expf(mrow[j] - mnew);
            lrow[j] = lrow[j] * corr + rs;
            mrow[j] = mnew;
#pragma unroll
            for (int t = 0; t < 4; ++t) acc[t][j] *= corr;
            int row = half ? 8 + j : j;
            pbuf[wslot][row * 32 + l16]      = f2bf(p0);
            pbuf[wslot][row * 32 + 16 + l16] = f2bf(p1);
        }

        // ---- read P back in A layout (DS in-order per wave: no barrier)
        v16bf pa;
        {
            const __bf16* pr = &pbuf[wslot][l16 * 32];
            v8bf lo = *(const v8bf*)(pr + half * 16);
            v8bf hi = *(const v8bf*)(pr + half * 16 + 8);
            pa = cat8(lo, hi);
        }

        // ---- acc += P(16x32) * V(32x64): 4 output tiles across DK
#pragma unroll
        for (int t = 0; t < 4; ++t) {
            const __bf16* vrow = Vb + (size_t)(t * 16 + l16) * SS + kc0;
            v16bf bv16 = *(const v16bf*)(vrow + half * 16);
            acc[t] = __builtin_amdgcn_wmma_f32_16x16x32_bf16(false, pa, false,
                                                             bv16, (short)0,
                                                             acc[t], false, false);
        }
    }

    // ---- normalize and write attn output as [B, S, H*DK] bf16
#pragma unroll
    for (int j = 0; j < 8; ++j) {
        int row = q0 + (half ? 8 + j : j);
        float inv = 1.f / lrow[j];
#pragma unroll
        for (int t = 0; t < 4; ++t) {
            int dk = t * 16 + l16;
            attn[((size_t)(b * SS + row)) * DD + h * DK + dk] = f2bf(acc[t][j] * inv);
        }
    }
}

// ---------------------------------------------------------------------------
extern "C" void kernel_launch(void* const* d_in, const int* in_sizes, int n_in,
                              void* d_out, int out_size, void* d_ws, size_t ws_size,
                              hipStream_t stream) {
    const float* qin = (const float*)d_in[0];
    const float* kin = (const float*)d_in[1];
    const float* vin = (const float*)d_in[2];
    // d_in[3] = mask: causal tril, applied analytically in attn_kernel
    const float* Wq = (const float*)d_in[4];
    const float* bq = (const float*)d_in[5];
    const float* Wk = (const float*)d_in[6];
    const float* bk = (const float*)d_in[7];
    const float* Wv = (const float*)d_in[8];
    const float* bv = (const float*)d_in[9];
    const float* Wo = (const float*)d_in[10];
    const float* bo = (const float*)d_in[11];
    float* out = (float*)d_out;

    const size_t nBSD = (size_t)BB * SS * DD;   // 4M elements
    const size_t nDD  = (size_t)DD * DD;        // 1M elements

    __bf16* ws = (__bf16*)d_ws;
    __bf16* qx   = ws;
    __bf16* kx   = ws + 1 * nBSD;
    __bf16* vx   = ws + 2 * nBSD;
    __bf16* wqb  = ws + 3 * nBSD;
    __bf16* wkb  = wqb + nDD;
    __bf16* wvb  = wkb + nDD;
    __bf16* wob  = wvb + nDD;
    __bf16* Qh   = wob + nDD;                   // [B,H,S,DK]
    __bf16* Kh   = Qh + nBSD;                   // [B,H,S,DK]
    __bf16* Vt   = Kh + nBSD;                   // [B,H,DK,S]
    __bf16* attn = Vt + nBSD;                   // [B,S,D]

    // 1) convert inputs/weights to bf16
    {
        int nb = (int)((nBSD + 255) / 256);
        cvt_f32_to_bf16<<<nb, 256, 0, stream>>>(qin, qx, (int)nBSD);
        cvt_f32_to_bf16<<<nb, 256, 0, stream>>>(kin, kx, (int)nBSD);
        cvt_f32_to_bf16<<<nb, 256, 0, stream>>>(vin, vx, (int)nBSD);
        int nw = (int)((nDD + 255) / 256);
        cvt_f32_to_bf16<<<nw, 256, 0, stream>>>(Wq, wqb, (int)nDD);
        cvt_f32_to_bf16<<<nw, 256, 0, stream>>>(Wk, wkb, (int)nDD);
        cvt_f32_to_bf16<<<nw, 256, 0, stream>>>(Wv, wvb, (int)nDD);
        cvt_f32_to_bf16<<<nw, 256, 0, stream>>>(Wo, wob, (int)nDD);
    }

    // 2) projections: (BS/32)*(DD/64) = 2048 waves -> 512 blocks of 4 waves
    const int projBlocks = (BS / 32) * (DD / 64) / 4;   // 512
    proj_gemm_kernel<<<projBlocks, 128, 0, stream>>>(qx, wqb, bq, Qh, 0);
    proj_gemm_kernel<<<projBlocks, 128, 0, stream>>>(kx, wkb, bk, Kh, 0);
    proj_gemm_kernel<<<projBlocks, 128, 0, stream>>>(vx, wvb, bv, Vt, 1);

    // 3) causal flash attention: B*H*(S/16) = 4096 waves -> 1024 blocks
    attn_kernel<<<(BB * HH * (SS / 16)) / 4, 128, 0, stream>>>(Qh, Kh, Vt, attn);

    // 4) output projection (f32 out)
    out_gemm_kernel<<<projBlocks, 128, 0, stream>>>(attn, wob, bo, out);
}
